// ScaledDotProductAttention_22187801051538
// MI455X (gfx1250) — compile-verified, matrix operand order checked by reading, not run on hardware
//
#include <hip/hip_runtime.h>
#include <math.h>

// Tanh-attention (no softmax; the reference discards the mask's effect):
//   scores = (Q K^T) / sqrt(24);  attn = tanh(0.1 * scores);  O = attn V
// B=2, H=8, S=2048, D=24 (fixed).
//
// f16 WMMA 16x16x32 / f32 accumulate, flash-style k-loop. Scores computed
// transposed (S^T = K Q^T) so matmul#1's C-layout feeds matmul#2's B-operand
// layout with purely per-lane elementwise tanh+pack (no shuffles).
// alpha/sqrt(24) is folded into the f16 Q operand, so the activation is a
// single hardware V_TANH_F32 per score element (gfx1250 TRANS op).
// K / V^T chunks are cooperatively staged into LDS as f16 tiles; the d=24..31
// zero padding is written once before the loop.

typedef _Float16 v16h __attribute__((ext_vector_type(16)));
typedef _Float16 v8h  __attribute__((ext_vector_type(8)));
typedef _Float16 v4h  __attribute__((ext_vector_type(4)));
typedef float    v8f  __attribute__((ext_vector_type(8)));
typedef float    v4f  __attribute__((ext_vector_type(4)));

#define SEQ    2048
#define DHEAD  24
#define NHEAD  16                 // B*H
#define WAVES_PER_BLOCK 8
#define QROWS_PER_BLOCK (WAVES_PER_BLOCK * 16)   // 128
#define QBLK_PER_HEAD   (SEQ / QROWS_PER_BLOCK)  // 16
#define LDS_STRIDE 40             // halves; 80B rows spread across banks

// Assemble a v16h A/B operand from two contiguous 8-half LDS chunks.
__device__ __forceinline__ v16h make_op(const _Float16* rowp, int base) {
  v8h lo = *(const v8h*)(rowp + base);
  v8h hi = *(const v8h*)(rowp + base + 16);
  return __builtin_shufflevector(lo, hi, 0, 1, 2, 3, 4, 5, 6, 7,
                                 8, 9, 10, 11, 12, 13, 14, 15);
}

__global__ __launch_bounds__(256) void
attn_tanh_wmma_f16(const float* __restrict__ Q,
                   const float* __restrict__ K,
                   const float* __restrict__ V,
                   float* __restrict__ O) {
  __shared__ _Float16 ksh[32 * LDS_STRIDE];  // [k_local][d]   d=24..31 zero
  __shared__ _Float16 vsh[32 * LDS_STRIDE];  // [d][k_local]   d=24..31 zero

  const int tid  = (int)threadIdx.x;
  const int wave = tid >> 5;
  const int lane = tid & 31;
  const int l16  = lane & 15;
  const int half = lane >> 4;
  const int base = half * 8;

  const int head = (int)blockIdx.x >> 4;          // 0..15
  const int qblk = (int)blockIdx.x & 15;
  const int q0   = qblk * QROWS_PER_BLOCK + wave * 16;

  const size_t hb = (size_t)head * SEQ * DHEAD;
  const float* __restrict__ Qh = Q + hb;
  const float* __restrict__ Kh = K + hb;
  const float* __restrict__ Vh = V + hb;
  float* __restrict__ Oh = O + hb;

  // ALPHA / sqrt(24): folded into the Q operand so the WMMA output is the
  // tanh argument directly.
  const float QPRESCALE = 0.1f * 0.20412414523193154f;

  // ---- One-time zero padding of the d = 24..31 regions (never rewritten).
  if (tid < 64) {
    const v4h z4 = {(_Float16)0.f, (_Float16)0.f, (_Float16)0.f, (_Float16)0.f};
    {   // ksh rows 0..31, halves 24..31  (two b64 stores per row)
      const int r  = tid >> 1;
      const int c  = 24 + (tid & 1) * 4;
      *(v4h*)(ksh + r * LDS_STRIDE + c) = z4;
    }
    {   // vsh rows (d) 24..31, halves 0..31
      const int r = 24 + (tid >> 3);
      const int c = (tid & 7) * 4;
      *(v4h*)(vsh + r * LDS_STRIDE + c) = z4;
    }
  }

  // ---- One-time load: B-operand of matmul#1 = pre-scaled Q^T. ----
  v16h bq;
  {
    const float* qrow = Qh + (size_t)(q0 + l16) * DHEAD;
#pragma unroll
    for (int i = 0; i < 8; ++i) {
      const int d0 = base + i;            // 0..15, always valid
      const int d1 = base + 16 + i;       // 16..31
      const int d1c = d1 < DHEAD ? d1 : 0;
      const float x1 = qrow[d1c];
      bq[i]     = (_Float16)(qrow[d0] * QPRESCALE);
      bq[i + 8] = (_Float16)((d1 < DHEAD ? x1 : 0.0f) * QPRESCALE);
    }
  }

  v8f acc0 = {0.f, 0.f, 0.f, 0.f, 0.f, 0.f, 0.f, 0.f};  // ctx^T, d = 0..15
  v8f acc1 = {0.f, 0.f, 0.f, 0.f, 0.f, 0.f, 0.f, 0.f};  // ctx^T, d = 16..23
  const v8f zero8 = {0.f, 0.f, 0.f, 0.f, 0.f, 0.f, 0.f, 0.f};

  // Staging split: 32 rows * 24 floats = 768 floats = 192 float4 chunks.
  // Waves 0..5 (threads 0..191) each move one float4 of K and of V.
  const int sr = tid / 6;        // row 0..31   (for tid < 192)
  const int sc = (tid % 6) * 4;  // col 0,4,..,20

  for (int kc = 0; kc < SEQ; kc += 32) {
    __syncthreads();   // protect LDS from previous iteration's readers
    if (tid < 192) {
      const size_t g = (size_t)(kc + sr) * DHEAD + sc;   // 16B-aligned
      const v4f kf = *(const v4f*)(Kh + g);
      const v4f vf = *(const v4f*)(Vh + g);
      v4h kh4;
      kh4[0] = (_Float16)kf[0]; kh4[1] = (_Float16)kf[1];
      kh4[2] = (_Float16)kf[2]; kh4[3] = (_Float16)kf[3];
      *(v4h*)(ksh + sr * LDS_STRIDE + sc) = kh4;          // one b64 store
#pragma unroll
      for (int i = 0; i < 4; ++i) {                        // transposed V
        vsh[(sc + i) * LDS_STRIDE + sr] = (_Float16)vf[i];
      }
    }
    __syncthreads();

    // ---- Matmul#1: m^T = (alpha*scale) * K Q^T, two 16-row k tiles. ----
    const v16h ak0 = make_op(ksh + (size_t)l16 * LDS_STRIDE, base);
    const v16h ak1 = make_op(ksh + (size_t)(16 + l16) * LDS_STRIDE, base);

    v8f c0 = __builtin_amdgcn_wmma_f32_16x16x32_f16(false, ak0, false, bq,
                                                    (short)0, zero8, false, false);
    v8f c1 = __builtin_amdgcn_wmma_f32_16x16x32_f16(false, ak1, false, bq,
                                                    (short)0, zero8, false, false);

    // ---- attn = tanh(m): one hardware V_TANH_F32 per element. ----
    v16h battn;
#pragma unroll
    for (int i = 0; i < 8; ++i) {
      battn[i]     = (_Float16)__builtin_amdgcn_tanhf(c0[i]);
      battn[i + 8] = (_Float16)__builtin_amdgcn_tanhf(c1[i]);
    }

    // ---- Matmul#2: ctx^T += V^T x attn^T (A = V^T tiles from LDS). ----
    const v16h av0 = make_op(vsh + (size_t)l16 * LDS_STRIDE, base);
    const v16h av1 = make_op(vsh + (size_t)(16 + l16) * LDS_STRIDE, base);

    acc0 = __builtin_amdgcn_wmma_f32_16x16x32_f16(false, av0, false, battn,
                                                  (short)0, acc0, false, false);
    acc1 = __builtin_amdgcn_wmma_f32_16x16x32_f16(false, av1, false, battn,
                                                  (short)0, acc1, false, false);
  }

  // ctx^T C-layout: lane: q = l16 ; acc0 VGPR i: d = i + half*8 ;
  //                 acc1 VGPR i: d = 16 + i + half*8 (store only d < 24).
  float* __restrict__ orow = Oh + (size_t)(q0 + l16) * DHEAD;
#pragma unroll
  for (int i = 0; i < 8; ++i) {
    orow[base + i] = acc0[i];
  }
  if (half == 0) {
#pragma unroll
    for (int i = 0; i < 8; ++i) {
      orow[16 + i] = acc1[i];
    }
  }
}

extern "C" void kernel_launch(void* const* d_in, const int* in_sizes, int n_in,
                              void* d_out, int out_size, void* d_ws, size_t ws_size,
                              hipStream_t stream) {
  (void)in_sizes; (void)n_in; (void)d_ws; (void)ws_size; (void)out_size;
  const float* Q = (const float*)d_in[0];
  const float* K = (const float*)d_in[1];
  const float* V = (const float*)d_in[2];
  // d_in[3] is the attention mask; the reference discards its effect.
  float* O = (float*)d_out;

  dim3 grid(NHEAD * QBLK_PER_HEAD);   // 256 blocks: (head, 128-row q slab)
  dim3 block(32 * WAVES_PER_BLOCK);   // 256 threads = 8 waves
  hipLaunchKernelGGL(attn_tanh_wmma_f16, grid, block, 0, stream, Q, K, V, O);
}
